// TransformerBlock_25872882991961
// MI455X (gfx1250) — matrix operand order repaired and checked by reading, not compiled, not run on hardware
//
#include <hip/hip_runtime.h>

typedef __attribute__((ext_vector_type(16))) __bf16 bf16x16;
typedef __attribute__((ext_vector_type(4)))  __bf16 bf16x4;
typedef __attribute__((ext_vector_type(8)))  float  f32x8;

union FragB16 { uint4 u[2]; bf16x16 v; };

#define WMMA_BF16(a, b, c) \
  __builtin_amdgcn_wmma_f32_16x16x32_bf16(false, (a), false, (b), (short)0, (c), false, false)

constexpr int Bn = 2, Sn = 2048, Dn = 1024, Hn = 16;
constexpr int Mn = Bn * Sn; // 4096 token rows

// Async copy global -> LDS (16B per lane), tracked by ASYNCcnt. No VGPR
// round-trip; LDS byte address = low 32 bits of the flat shared pointer.
__device__ __forceinline__ void async_copy16(const __bf16* __restrict__ g,
                                             const __bf16* __restrict__ l) {
  const unsigned lds = (unsigned)(uintptr_t)l;
  asm volatile("global_load_async_to_lds_b128 %0, %1, off"
               :: "v"(lds), "v"(g) : "memory");
}
__device__ __forceinline__ void async_wait8() {
  asm volatile("s_wait_asynccnt 0x8" ::: "memory");
}
__device__ __forceinline__ void async_wait0() {
  asm volatile("s_wait_asynccnt 0x0" ::: "memory");
}

// ---------------------------------------------------------------------------
// LayerNorm: one wave per row (D=1024), float4 loads, shfl reductions,
// bf16x4 (b64) stores feeding the WMMA GEMMs.
// ---------------------------------------------------------------------------
__global__ __launch_bounds__(256)
void ln_bf16_kernel(const float* __restrict__ x, const float* __restrict__ gamma,
                    const float* __restrict__ beta, __bf16* __restrict__ out, int nrows) {
  const int wave = threadIdx.x >> 5, lane = threadIdx.x & 31;
  const int row = blockIdx.x * 8 + wave;
  if (row >= nrows) return;
  const float4* xr = (const float4*)(x + (size_t)row * Dn);
  float4 v[8];
  float s = 0.f, s2 = 0.f;
#pragma unroll
  for (int i = 0; i < 8; ++i) {
    const float4 t = xr[lane + i * 32];
    v[i] = t;
    s  += t.x + t.y + t.z + t.w;
    s2 += t.x * t.x + t.y * t.y + t.z * t.z + t.w * t.w;
  }
#pragma unroll
  for (int m = 16; m >= 1; m >>= 1) {
    s  += __shfl_xor(s,  m, 32);
    s2 += __shfl_xor(s2, m, 32);
  }
  const float mu  = s * (1.f / Dn);
  const float var = s2 * (1.f / Dn) - mu * mu;
  const float rs  = rsqrtf(var + 1e-5f);
  const float4* g4 = (const float4*)gamma;
  const float4* b4 = (const float4*)beta;
  bf16x4* o4 = (bf16x4*)(out + (size_t)row * Dn);
#pragma unroll
  for (int i = 0; i < 8; ++i) {
    const int e = lane + i * 32;
    const float4 gv = g4[e], bv = b4[e];
    bf16x4 o;
    o[0] = (__bf16)(gv.x * ((v[i].x - mu) * rs) + bv.x);
    o[1] = (__bf16)(gv.y * ((v[i].y - mu) * rs) + bv.y);
    o[2] = (__bf16)(gv.z * ((v[i].z - mu) * rs) + bv.z);
    o[3] = (__bf16)(gv.w * ((v[i].w - mu) * rs) + bv.w);
    o4[e] = o;
  }
}

// ---------------------------------------------------------------------------
// Weight convert + transpose: W fp32 [K,N] -> Wt bf16 [N,K] so WMMA B-frags
// (lane = column n, K contiguous) become contiguous 32B loads.
// ---------------------------------------------------------------------------
__global__ __launch_bounds__(256)
void wtrans_kernel(const float* __restrict__ W, __bf16* __restrict__ Wt, int K, int N) {
  __shared__ float tile[32][33];
  const int n0 = blockIdx.x * 32, k0 = blockIdx.y * 32;
#pragma unroll
  for (int j = 0; j < 4; ++j)
    tile[threadIdx.y + j * 8][threadIdx.x] =
        W[(size_t)(k0 + threadIdx.y + j * 8) * N + n0 + threadIdx.x];
  __syncthreads();
#pragma unroll
  for (int j = 0; j < 4; ++j)
    Wt[(size_t)(n0 + threadIdx.y + j * 8) * K + k0 + threadIdx.x] =
        (__bf16)tile[threadIdx.x][threadIdx.y + j * 8];
}

// ---------------------------------------------------------------------------
// bf16 GEMM: C[M,N] = A[M,K] (row-major bf16) x Bt[N,K] (bf16 = B^T).
// 256 thr = 8 waves, 128x128 block tile, wave = 64x32 (8 v8f accumulators).
// K stepped by 64 with DOUBLE-BUFFERED LDS filled by async global->LDS copies
// (ASYNCcnt), so the next tile streams in while WMMAs consume the current one.
// LDS rows padded to 144B (16B-aligned, 36-bank stride: conflict-free).
// Epilogue modes:
//   0: fp32 row-major     1: bf16 [B,H,S,DH] (Q/K, scale folded)
//   2: fp32 + residual    3: bf16 [B,H,DH,S] (V transposed for PV WMMA)
//   4: bf16 row-major
// ---------------------------------------------------------------------------
__global__ __launch_bounds__(256)
void gemm_bf16_kernel(const __bf16* __restrict__ A, const __bf16* __restrict__ Bt,
                      int M, int N, int K, int mode, float scale,
                      const float* __restrict__ resid, float* __restrict__ outf,
                      __bf16* __restrict__ outb) {
  constexpr int LROW = 72;                         // bf16 elems/row (64 + 8 pad)
  __shared__ __align__(16) __bf16 As[2][128 * LROW];
  __shared__ __align__(16) __bf16 Bs[2][128 * LROW];
  const int lane = threadIdx.x & 31, wave = threadIdx.x >> 5;
  const int wm = wave >> 2, wn = wave & 3;         // 2 (M) x 4 (N) wave grid
  const int lm = lane & 15, half = lane >> 4;
  const int bm = blockIdx.y * 128, bn = blockIdx.x * 128;

  const f32x8 vzero = {0.f, 0.f, 0.f, 0.f, 0.f, 0.f, 0.f, 0.f};
  f32x8 acc[4][2];
#pragma unroll
  for (int mt = 0; mt < 4; ++mt)
#pragma unroll
    for (int nt = 0; nt < 2; ++nt) acc[mt][nt] = vzero;

  const int t = threadIdx.x;
  // Fill one 128x64 double-buffer slot for A and B: 1024 16B segments each,
  // 4 per thread per matrix -> 8 async instructions per wave per tile.
  auto issue_tile = [&](int kb, int buf) {
#pragma unroll
    for (int j = 0; j < 4; ++j) {
      const int sseg = t + 256 * j;                // 0..1023
      const int r = sseg >> 3, sc = (sseg & 7) * 8;
      async_copy16(A  + (size_t)(bm + r) * K + kb + sc, &As[buf][r * LROW + sc]);
      async_copy16(Bt + (size_t)(bn + r) * K + kb + sc, &Bs[buf][r * LROW + sc]);
    }
  };

  issue_tile(0, 0);
  int buf = 0;
  for (int kb = 0; kb < K; kb += 64) {
    const bool has_next = (kb + 64 < K);
    if (has_next) issue_tile(kb + 64, buf ^ 1);
    if (kb + 128 < K) {  // warm L2 two tiles ahead (partial, rows 0..31)
      const int sseg = t, r = sseg >> 3, sc = (sseg & 7) * 8;
      __builtin_prefetch(A  + (size_t)(bm + r) * K + kb + 128 + sc, 0, 1);
      __builtin_prefetch(Bt + (size_t)(bn + r) * K + kb + 128 + sc, 0, 1);
    }
    if (has_next) async_wait8(); else async_wait0();
    __syncthreads();                               // current buffer visible

#pragma unroll
    for (int kc = 0; kc < 64; kc += 32) {
      FragB16 bf0, bf1;
      {
        int n = wn * 32 + lm;
        const __bf16* bp = &Bs[buf][n * LROW + kc + 16 * half];
        bf0.u[0] = *(const uint4*)bp;
        bf0.u[1] = *(const uint4*)(bp + 8);
        bp += 16 * LROW;
        bf1.u[0] = *(const uint4*)bp;
        bf1.u[1] = *(const uint4*)(bp + 8);
      }
#pragma unroll
      for (int mt = 0; mt < 4; ++mt) {
        FragB16 af;
        const int m = wm * 64 + mt * 16 + lm;
        const __bf16* ap = &As[buf][m * LROW + kc + 16 * half];
        af.u[0] = *(const uint4*)ap;
        af.u[1] = *(const uint4*)(ap + 8);
        acc[mt][0] = WMMA_BF16(af.v, bf0.v, acc[mt][0]);
        acc[mt][1] = WMMA_BF16(af.v, bf1.v, acc[mt][1]);
      }
    }
    __syncthreads();                               // done reading before refill
    buf ^= 1;
  }

#pragma unroll
  for (int mt = 0; mt < 4; ++mt)
#pragma unroll
    for (int nt = 0; nt < 2; ++nt)
#pragma unroll
      for (int g = 0; g < 8; ++g) {
        const int row = bm + wm * 64 + mt * 16 + g + 8 * half;
        const int col = bn + wn * 32 + nt * 16 + lm;
        const float vacc = acc[mt][nt][g];
        if (mode == 2) {
          const size_t idx = (size_t)row * N + col;
          outf[idx] = resid[idx] + vacc;
        } else if (mode == 4) {
          outb[(size_t)row * N + col] = (__bf16)vacc;
        } else if (mode == 1) {
          const int b = row / Sn, s = row - b * Sn;
          const int h = col >> 6, dh = col & 63;
          outb[(((size_t)(b * Hn + h) * Sn + s) << 6) + dh] = (__bf16)(vacc * scale);
        } else if (mode == 3) {
          const int b = row / Sn, s = row - b * Sn;
          const int h = col >> 6, dh = col & 63;
          outb[((size_t)(b * Hn + h) * 64 + dh) * Sn + s] = (__bf16)vacc;
        } else {
          outf[(size_t)row * N + col] = vacc;
        }
      }
}

// ---------------------------------------------------------------------------
// Flash attention (causal): 4 waves/block, one 16-query tile per wave.
// q,k: bf16 [B*H, S, 64] (q pre-scaled by 1/8); vt: bf16 [B*H, 64, S].
// Online softmax: row stats are uniform per lane-half -> 16-lane shfl_xor
// reductions only. P re-swizzled C->A layout via per-wave 1KB LDS scratch.
// Diagonal blocks skip the fully-masked upper 16-key tile (wave-uniform).
// ---------------------------------------------------------------------------
__global__ __launch_bounds__(128)
void attn_kernel(const __bf16* __restrict__ q, const __bf16* __restrict__ k,
                 const __bf16* __restrict__ vt, const int* __restrict__ amask,
                 __bf16* __restrict__ out) {
  __shared__ __align__(16) __bf16 pbuf[4][16 * 32];
  const int wave = threadIdx.x >> 5, lane = threadIdx.x & 31;
  const int lm = lane & 15, half = lane >> 4;
  const int qtile = blockIdx.x * 64 + wave * 16;
  const int bh = blockIdx.y;
  const int b = bh >> 4, h = bh & 15;

  // Q A-fragments (16 x 64 as two k=32 chunks), contiguous k order (k = 16*half + s), same as the K B-frags.
  FragB16 af[2];
  const __bf16* qrow = q + ((size_t)bh * Sn + qtile + lm) * 64;
#pragma unroll
  for (int c = 0; c < 2; ++c) {
    af[c].u[0] = *(const uint4*)(qrow + c * 32 + 16 * half);
    af[c].u[1] = *(const uint4*)(qrow + c * 32 + 16 * half + 8);
  }

  const f32x8 vzero = {0.f, 0.f, 0.f, 0.f, 0.f, 0.f, 0.f, 0.f};
  f32x8 o[4];
#pragma unroll
  for (int nt = 0; nt < 4; ++nt) o[nt] = vzero;
  float mrow[8], lrow[8];
#pragma unroll
  for (int g = 0; g < 8; ++g) { mrow[g] = -1e30f; lrow[g] = 0.f; }

  for (int kb = 0; kb < qtile + 16; kb += 32) {
    float sv[2][8];
#pragma unroll
    for (int kt = 0; kt < 2; ++kt) {
      if (kt == 1 && kb >= qtile) {                // fully-masked diag tile
#pragma unroll
        for (int g = 0; g < 8; ++g) sv[1][g] = -1e30f;
        continue;
      }
      f32x8 sc = vzero;
      const __bf16* krow = k + ((size_t)bh * Sn + kb + kt * 16 + lm) * 64;
      __builtin_prefetch(krow + 32 * 64, 0, 1);    // next key block, same lane
#pragma unroll
      for (int c = 0; c < 2; ++c) {
        FragB16 bf;
        bf.u[0] = *(const uint4*)(krow + c * 32 + 16 * half);
        bf.u[1] = *(const uint4*)(krow + c * 32 + 16 * half + 8);
        sc = WMMA_BF16(af[c].v, bf.v, sc);
      }
      const int key = kb + kt * 16 + lm;
      const int am = amask[b * Sn + key];
#pragma unroll
      for (int g = 0; g < 8; ++g) {
        const int qi = qtile + g + 8 * half;
        sv[kt][g] = (key > qi || am == 0) ? -1e30f : sc[g];
      }
    }

    float alpha[8];
#pragma unroll
    for (int g = 0; g < 8; ++g) {
      float tmx = fmaxf(sv[0][g], sv[1][g]);
#pragma unroll
      for (int m = 8; m >= 1; m >>= 1) tmx = fmaxf(tmx, __shfl_xor(tmx, m, 32));
      const float mnew = fmaxf(mrow[g], tmx);
      alpha[g] = __expf(mrow[g] - mnew);
      mrow[g] = mnew;
      const float p0 = __expf(sv[0][g] - mnew);
      const float p1 = __expf(sv[1][g] - mnew);
      float rsum = p0 + p1;
#pragma unroll
      for (int m = 8; m >= 1; m >>= 1) rsum += __shfl_xor(rsum, m, 32);
      lrow[g] = lrow[g] * alpha[g] + rsum;
      pbuf[wave][(g + 8 * half) * 32 + lm]      = (__bf16)p0;
      pbuf[wave][(g + 8 * half) * 32 + 16 + lm] = (__bf16)p1;
    }
#pragma unroll
    for (int nt = 0; nt < 4; ++nt)
#pragma unroll
      for (int g = 0; g < 8; ++g) o[nt][g] *= alpha[g];

    asm volatile("s_wait_dscnt 0x0" ::: "memory"); // DS RAW fence (same wave)

    FragB16 pf;  // P as A-fragment (16 queries x 32 keys)
    pf.u[0] = *(const uint4*)&pbuf[wave][lm * 32 + 16 * half];
    pf.u[1] = *(const uint4*)&pbuf[wave][lm * 32 + 16 * half + 8];

#pragma unroll
    for (int nt = 0; nt < 4; ++nt) {
      const __bf16* vrow = vt + ((size_t)bh * 64 + nt * 16 + lm) * Sn + kb + 16 * half;
      __builtin_prefetch(vrow + 32, 0, 1);         // next key block (contig in S)
      FragB16 bf;
      bf.u[0] = *(const uint4*)vrow;
      bf.u[1] = *(const uint4*)(vrow + 8);
      o[nt] = WMMA_BF16(pf.v, bf.v, o[nt]);
    }
  }

#pragma unroll
  for (int g = 0; g < 8; ++g) {
    const float inv = 1.f / fmaxf(lrow[g], 1e-20f);
    const int qi = qtile + g + 8 * half;
    __bf16* orow = out + ((size_t)(b * Sn + qi)) * Dn + h * 64;
#pragma unroll
    for (int nt = 0; nt < 4; ++nt) orow[nt * 16 + lm] = (__bf16)(o[nt][g] * inv);
  }
}

// ---------------------------------------------------------------------------
// GLU: glu[m,n] = h[m,n] * sigmoid(h[m, 2048+n]); 4-wide bf16 vectors.
// ---------------------------------------------------------------------------
__global__ __launch_bounds__(256)
void glu_kernel(const __bf16* __restrict__ h, __bf16* __restrict__ g, size_t total4) {
  const size_t i = (size_t)blockIdx.x * blockDim.x + threadIdx.x;
  if (i >= total4) return;
  const size_t m = i >> 9;                 // 512 bf16x4 per output row
  const int n = (int)(i & 511) * 4;
  const bf16x4 l = *(const bf16x4*)&h[m * 4096 + n];
  const bf16x4 r = *(const bf16x4*)&h[m * 4096 + 2048 + n];
  bf16x4 o;
#pragma unroll
  for (int j = 0; j < 4; ++j) {
    const float lf = (float)l[j], rf = (float)r[j];
    o[j] = (__bf16)(lf / (1.f + __expf(-rf)));
  }
  *(bf16x4*)&g[m * 2048 + n] = o;
}

// ---------------------------------------------------------------------------
extern "C" void kernel_launch(void* const* d_in, const int* in_sizes, int n_in,
                              void* d_out, int out_size, void* d_ws, size_t ws_size,
                              hipStream_t stream) {
  (void)in_sizes; (void)n_in; (void)out_size; (void)ws_size;
  const float* x  = (const float*)d_in[0];
  const int*   am = (const int*)d_in[1];
  const float* Wq = (const float*)d_in[2];
  const float* Wk = (const float*)d_in[3];
  const float* Wv = (const float*)d_in[4];
  const float* Wo = (const float*)d_in[5];
  const float* W1 = (const float*)d_in[6];
  const float* W2 = (const float*)d_in[7];
  const float* g1 = (const float*)d_in[8];
  const float* b1 = (const float*)d_in[9];
  const float* g2 = (const float*)d_in[10];
  const float* b2 = (const float*)d_in[11];
  float* out = (float*)d_out;

  char* ws = (char*)d_ws;
  size_t off = 0;
  auto carve = [&](size_t bytes) -> void* {
    void* p = ws + off;
    off += (bytes + 255) & ~(size_t)255;
    return p;
  };
  __bf16* ln1  = (__bf16*)carve((size_t)Mn * Dn * 2);
  __bf16* wqt  = (__bf16*)carve((size_t)Dn * Dn * 2);
  __bf16* wkt  = (__bf16*)carve((size_t)Dn * Dn * 2);
  __bf16* wvt  = (__bf16*)carve((size_t)Dn * Dn * 2);
  __bf16* wot  = (__bf16*)carve((size_t)Dn * Dn * 2);
  __bf16* w1t  = (__bf16*)carve((size_t)Dn * 4 * Dn * 2);
  __bf16* w2t  = (__bf16*)carve((size_t)2 * Dn * Dn * 2);
  __bf16* qb   = (__bf16*)carve((size_t)Mn * Dn * 2);
  __bf16* kbuf = (__bf16*)carve((size_t)Mn * Dn * 2);
  __bf16* vtb  = (__bf16*)carve((size_t)Mn * Dn * 2);
  __bf16* atb  = (__bf16*)carve((size_t)Mn * Dn * 2);
  float*  x1   = (float*) carve((size_t)Mn * Dn * 4);
  __bf16* ln2  = (__bf16*)carve((size_t)Mn * Dn * 2);
  __bf16* hb   = (__bf16*)carve((size_t)Mn * 4 * Dn * 2);
  __bf16* glub = (__bf16*)carve((size_t)Mn * 2 * Dn * 2);

  const dim3 tb32(32, 8);
  // weights -> bf16, transposed
  wtrans_kernel<<<dim3(Dn / 32, Dn / 32), tb32, 0, stream>>>(Wq, wqt, Dn, Dn);
  wtrans_kernel<<<dim3(Dn / 32, Dn / 32), tb32, 0, stream>>>(Wk, wkt, Dn, Dn);
  wtrans_kernel<<<dim3(Dn / 32, Dn / 32), tb32, 0, stream>>>(Wv, wvt, Dn, Dn);
  wtrans_kernel<<<dim3(Dn / 32, Dn / 32), tb32, 0, stream>>>(Wo, wot, Dn, Dn);
  wtrans_kernel<<<dim3(4 * Dn / 32, Dn / 32), tb32, 0, stream>>>(W1, w1t, Dn, 4 * Dn);
  wtrans_kernel<<<dim3(Dn / 32, 2 * Dn / 32), tb32, 0, stream>>>(W2, w2t, 2 * Dn, Dn);

  // LN1
  ln_bf16_kernel<<<Mn / 8, 256, 0, stream>>>(x, g1, b1, ln1, Mn);

  // QKV projections (Q scaled by 1/sqrt(DH)=0.125; V stored transposed)
  const dim3 gq(Dn / 128, Mn / 128);
  gemm_bf16_kernel<<<gq, 256, 0, stream>>>(ln1, wqt, Mn, Dn, Dn, 1, 0.125f, nullptr, nullptr, qb);
  gemm_bf16_kernel<<<gq, 256, 0, stream>>>(ln1, wkt, Mn, Dn, Dn, 1, 1.0f,   nullptr, nullptr, kbuf);
  gemm_bf16_kernel<<<gq, 256, 0, stream>>>(ln1, wvt, Mn, Dn, Dn, 3, 1.0f,   nullptr, nullptr, vtb);

  // Causal flash attention
  attn_kernel<<<dim3(Sn / 64, Bn * Hn), 128, 0, stream>>>(qb, kbuf, vtb, am, atb);

  // Output projection + residual -> x1 (fp32)
  gemm_bf16_kernel<<<gq, 256, 0, stream>>>(atb, wot, Mn, Dn, Dn, 2, 1.0f, x, x1, nullptr);

  // LN2
  ln_bf16_kernel<<<Mn / 8, 256, 0, stream>>>(x1, g2, b2, ln2, Mn);

  // MLP up (N = 4D), GLU, MLP down + residual -> out
  gemm_bf16_kernel<<<dim3(4 * Dn / 128, Mn / 128), 256, 0, stream>>>(
      ln2, w1t, Mn, 4 * Dn, Dn, 4, 1.0f, nullptr, nullptr, hb);
  glu_kernel<<<(unsigned)(((size_t)Mn * 2 * Dn / 4 + 255) / 256), 256, 0, stream>>>(
      hb, glub, (size_t)Mn * 2 * Dn / 4);
  gemm_bf16_kernel<<<gq, 256, 0, stream>>>(glub, w2t, Mn, Dn, 2 * Dn, 2, 1.0f, x1, out, nullptr);
}
